// ModRnn_8409545965982
// MI455X (gfx1250) — matrix operand order repaired and checked by reading, not compiled, not run on hardware
//
#include <hip/hip_runtime.h>

// CDNA5 / gfx1250, wave32
typedef __attribute__((ext_vector_type(16))) __bf16 v16bf;
typedef __attribute__((ext_vector_type(8)))  __bf16 v8bf;
typedef __attribute__((ext_vector_type(8)))  float  v8f;

#define NHID    256
#define BATCH   128
#define TLEN    2048
#define MROWS   16      // batch rows per workgroup (one WMMA M-tile)
#define HSTRIDE 264     // padded bf16 row stride in LDS (528B, 16B aligned, bank-skewed)

// Prefer CDNA5 hardware V_TANH_F32 (single TRANS op, co-executes with WMMA).
// Fallback: branch-free tanh via v_exp_f32 + v_rcp_f32.
__device__ __forceinline__ float fast_tanh(float v) {
#if __has_builtin(__builtin_amdgcn_tanhf)
    return __builtin_amdgcn_tanhf(v);
#else
    float e = __builtin_amdgcn_exp2f(v * 2.8853900817779268f); // 2*log2(e)
    return 1.0f - 2.0f * __builtin_amdgcn_rcpf(e + 1.0f);
#endif
}

__launch_bounds__(256, 1)
__global__ void rnn_wmma_kernel(const float* __restrict__ x,
                                const float* __restrict__ W_ih,
                                const float* __restrict__ W_hh,
                                const float* __restrict__ b_ih,
                                const float* __restrict__ b_hh,
                                const float* __restrict__ W_out,
                                const float* __restrict__ b_out,
                                float* __restrict__ y)
{
    // Double-buffered by step parity -> exactly ONE barrier per time step.
    __shared__ __bf16 hbuf[2][MROWS * HSTRIDE];  // h_t tile, bf16
    __shared__ float  xs[2][MROWS];              // x for current step
    __shared__ float  ywave[2][8][MROWS];        // per-wave y partials

    const int tid  = threadIdx.x;
    const int w    = tid >> 5;        // wave id 0..7
    const int lane = tid & 31;
    const int l16  = lane & 15;
    const int hi   = lane >> 4;       // which 16-lane half
    const int hi8  = hi * 8;
    const int rowbase = blockIdx.x * MROWS;   // 8 blocks x 16 rows = 128

    // Two hidden columns owned by this lane (two 16x16 N tiles per wave)
    const int n0 = w * 32 + l16;
    const int n1 = n0 + 16;

    const float wih0 = W_ih[n0],  wih1 = W_ih[n1];
    const float bia0 = b_ih[n0] + b_hh[n0];
    const float bia1 = b_ih[n1] + b_hh[n1];
    const float wo0  = W_out[n0], wo1  = W_out[n1];
    const float bo   = b_out[0];

    // ---- One-time: B fragments (W_hh^T) -> bf16, resident in VGPRs ----
    // B layout (16-bit, 32x16): lane -> N = l16; element e -> K = k0 + 16*hi + e
    v16bf Bfrag[2][8];
    const int khalf = hi * 16;
#pragma unroll
    for (int t = 0; t < 2; ++t) {
        const float* wrow = W_hh + (size_t)((t == 0) ? n0 : n1) * NHID;
#pragma unroll
        for (int kc = 0; kc < 8; ++kc) {
            const int kb = kc * 32 + khalf;
#pragma unroll
            for (int e = 0; e < 16; ++e)
                Bfrag[t][kc][e] = (__bf16)wrow[kb + e];
        }
    }

    // h(0) = 0 in parity-0 buffer; prefetch x(t=0)
    for (int i = tid; i < MROWS * HSTRIDE; i += 256)
        hbuf[0][i] = (__bf16)0.f;
    if (tid < MROWS) xs[0][tid] = x[(size_t)(rowbase + tid) * TLEN + 0];
    __syncthreads();

    const int m = l16;  // A-matrix row for this lane

    for (int t = 0; t < TLEN; ++t) {
        const int cur = t & 1;
        const int nxt = cur ^ 1;

        // ---- read A fragments (h_t) + x_t from LDS (parity cur) ----
        // A layout (16-bit, 16x32): lane -> M = l16;
        //   element e -> K = k0 + 16*hi + e
        v16bf Afrag[8];
#pragma unroll
        for (int kc = 0; kc < 8; ++kc) {
            const __bf16* base = &hbuf[cur][m * HSTRIDE + kc * 32 + hi * 16];
            v8bf lo  = *(const v8bf*)(base);
            v8bf hiv = *(const v8bf*)(base + 8);
#pragma unroll
            for (int e = 0; e < 8; ++e) { Afrag[kc][e] = lo[e]; Afrag[kc][8 + e] = hiv[e]; }
        }
        float x8[8];
#pragma unroll
        for (int r = 0; r < 8; ++r) x8[r] = xs[cur][r + hi8];

        // ---- C init = x_t * W_ih + (b_ih + b_hh), then WMMA over K ----
        v8f acc0, acc1;
#pragma unroll
        for (int r = 0; r < 8; ++r) {
            acc0[r] = x8[r] * wih0 + bia0;
            acc1[r] = x8[r] * wih1 + bia1;
        }
#pragma unroll
        for (int kc = 0; kc < 8; ++kc) {
            acc0 = __builtin_amdgcn_wmma_f32_16x16x32_bf16(
                false, Afrag[kc], false, Bfrag[0][kc], (short)0, acc0, false, false);
            acc1 = __builtin_amdgcn_wmma_f32_16x16x32_bf16(
                false, Afrag[kc], false, Bfrag[1][kc], (short)0, acc1, false, false);
        }

        // ---- tanh (hw v_tanh_f32), write h_{t+1} to parity nxt, y partials ----
        float p[8];
#pragma unroll
        for (int r = 0; r < 8; ++r) {
            float h0v = fast_tanh(acc0[r]);
            float h1v = fast_tanh(acc1[r]);
            const int row = r + hi8;   // C layout: M = r + 8*hi, N = lane column
            hbuf[nxt][row * HSTRIDE + n0] = (__bf16)h0v;
            hbuf[nxt][row * HSTRIDE + n1] = (__bf16)h1v;
            p[r] = h0v * wo0 + h1v * wo1;
        }
        // xor-butterfly reduce over each 16-lane half (rows 0..7 / 8..15)
#pragma unroll
        for (int r = 0; r < 8; ++r) {
#pragma unroll
            for (int mask = 1; mask < 16; mask <<= 1)
                p[r] += __shfl_xor(p[r], mask, 32);
            if (l16 == r) ywave[cur][w][r + hi8] = p[r];
        }
        if (tid < MROWS && (t + 1) < TLEN)
            xs[nxt][tid] = x[(size_t)(rowbase + tid) * TLEN + (t + 1)];

        __syncthreads();   // h_{t+1}, ywave(t), x_{t+1} visible

        if (tid < MROWS) {
            float s = 0.f;
#pragma unroll
            for (int ww = 0; ww < 8; ++ww) s += ywave[cur][ww][tid];
            y[(size_t)(rowbase + tid) * TLEN + t] = s + bo;
        }
    }
}

extern "C" void kernel_launch(void* const* d_in, const int* in_sizes, int n_in,
                              void* d_out, int out_size, void* d_ws, size_t ws_size,
                              hipStream_t stream) {
    const float* x     = (const float*)d_in[0];
    const float* W_ih  = (const float*)d_in[1];
    const float* W_hh  = (const float*)d_in[2];
    const float* b_ih  = (const float*)d_in[3];
    const float* b_hh  = (const float*)d_in[4];
    const float* W_out = (const float*)d_in[5];
    const float* b_out = (const float*)d_in[6];
    float* y = (float*)d_out;

    rnn_wmma_kernel<<<BATCH / MROWS, 256, 0, stream>>>(
        x, W_ih, W_hh, b_ih, b_hh, W_out, b_out, y);
}
